// Encoder_54107997995094
// MI455X (gfx1250) — compile-verified, matrix operand order checked by reading, not run on hardware
//
#include <hip/hip_runtime.h>
#include <hip/hip_bf16.h>

// Problem sizes (fixed by the reference)
#define S_ 512
#define B_ 64
#define I_ 512
#define H_ 1024

typedef __attribute__((ext_vector_type(16))) __bf16 v16bf;
typedef __attribute__((ext_vector_type(8)))  __bf16 v8bf;
typedef __attribute__((ext_vector_type(8)))  float  v8f;
typedef __attribute__((ext_vector_type(4)))  int    v4i;   // native 16B vector

#define LDK 40    // padded LDS row stride (bf16) for 32-wide K chunks (zgemm)
#define WST 1032  // padded LDS row stride (bf16) for full-H rows (persistent)

// Workspace layout (bytes)
#define XB_OFF   0u                               // x bf16:   32 MB
#define WIH_OFF  (XB_OFF  + (size_t)S_*B_*I_*2)   // W_ih bf16: 1 MB
#define WHH_OFF  (WIH_OFF + (size_t)H_*I_*2)      // W_hh bf16: 2 MB
#define BIAS_OFF (WHH_OFF + (size_t)H_*H_*2)      // bias f32:  4 KB
#define HB_OFF   (BIAS_OFF + (size_t)H_*4)        // 4 * (B*H) bf16 ping-pong h
#define BAR_OFF  (HB_OFF  + (size_t)4*B_*H_*2)    // 2 barrier counters (+pad)
#define Z_OFF    (BAR_OFF + (size_t)256)          // Z f32: 128 MB

// ---------------------------------------------------------------------------
// CDNA5 async global->LDS copy (ASYNCcnt path), guarded so compile never breaks
// ---------------------------------------------------------------------------
#if __has_builtin(__builtin_amdgcn_global_load_async_to_lds_b128)
#define HAVE_ASYNC_LDS 1
typedef __attribute__((address_space(1))) v4i g_v4i;  // AS1 = global
typedef __attribute__((address_space(3))) v4i l_v4i;  // AS3 = LDS
#endif

__device__ __forceinline__ void async_cp16(const void* g, void* l) {
#if defined(HAVE_ASYNC_LDS)
    __builtin_amdgcn_global_load_async_to_lds_b128(
        (g_v4i*)g, (l_v4i*)l, 0, 0);
#else
    *(v4i*)l = *(const v4i*)g;
#endif
}

__device__ __forceinline__ void async_wait0() {
#if defined(HAVE_ASYNC_LDS)
#if __has_builtin(__builtin_amdgcn_s_wait_asynccnt)
    __builtin_amdgcn_s_wait_asynccnt(0);
#else
    asm volatile("s_wait_asynccnt 0x0" ::: "memory");
#endif
#endif
}

union FragU { v16bf v; v8bf h[2]; };

// 16x32 bf16 operand fragment per ISA 7.12.2:
// lane (l&15) = row; K chunks [8*(l>>4), +8) and [16+8*(l>>4), +8).
__device__ __forceinline__ v16bf load_frag(const __bf16* row, int ksel) {
    FragU u;
    u.h[0] = *(const v8bf*)(row + ksel);
    u.h[1] = *(const v8bf*)(row + 16 + ksel);
    return u.v;
}

__device__ __forceinline__ v8f wmma_bf16(v16bf a, v16bf b, v8f c) {
    return __builtin_amdgcn_wmma_f32_16x16x32_bf16(
        false, a, false, b, (short)0, c, false, false);
}

// ---------------------------------------------------------------------------
// prep: f32 -> bf16 conversions, bias fuse, zero h buffers + barrier counters
// ---------------------------------------------------------------------------
__global__ void __launch_bounds__(256) prep_kernel(
    const float* __restrict__ x,   const float* __restrict__ Wih,
    const float* __restrict__ Whh, const float* __restrict__ bih,
    const float* __restrict__ bhh,
    __bf16* __restrict__ xb, __bf16* __restrict__ wihb, __bf16* __restrict__ whhb,
    float* __restrict__ bias, __bf16* __restrict__ hinit,
    unsigned* __restrict__ bar)
{
    const size_t i0 = (size_t)blockIdx.x * blockDim.x + threadIdx.x;
    const size_t st = (size_t)gridDim.x * blockDim.x;
    for (size_t i = i0; i < (size_t)S_ * B_ * I_; i += st) xb[i]   = (__bf16)x[i];
    for (size_t i = i0; i < (size_t)H_ * I_;      i += st) wihb[i] = (__bf16)Wih[i];
    for (size_t i = i0; i < (size_t)H_ * H_;      i += st) whhb[i] = (__bf16)Whh[i];
    for (size_t i = i0; i < (size_t)H_;           i += st) bias[i] = bih[i] + bhh[i];
    for (size_t i = i0; i < (size_t)4 * B_ * H_;  i += st) hinit[i] = (__bf16)0.0f;
    for (size_t i = i0; i < 2;                    i += st) bar[i] = 0u;
}

// ---------------------------------------------------------------------------
// zgemm: Z[m=t*B+b][n] = sum_k xb[m][k] * wihb[n][k] + bias[n]
// WG tile 128x128, waves 2x4, each wave 64x32 (4x2 WMMA tiles).
// ---------------------------------------------------------------------------
__global__ void __launch_bounds__(256) zgemm_kernel(
    const __bf16* __restrict__ xb, const __bf16* __restrict__ wih,
    const float* __restrict__ bias, float* __restrict__ Z)
{
    __shared__ alignas(16) __bf16 Al[128 * LDK];
    __shared__ alignas(16) __bf16 Bl[128 * LDK];

    const int m0   = blockIdx.x * 128;
    const int n0   = blockIdx.y * 128;
    const int tid  = threadIdx.x;
    const int lane = tid & 31, wave = tid >> 5;
    const int wm = (wave >> 2) * 64;
    const int wn = (wave & 3) * 32;
    const int ksel = (lane >> 4) * 8;
    const int rsel = lane & 15;

    v8f acc[4][2] = {};

    for (int k0 = 0; k0 < I_; k0 += 32) {
        __syncthreads();
        // A tile: 128 rows x 32 bf16 = 512 16B chunks; 2/thread, unrolled
        #pragma unroll
        for (int it = 0; it < 2; ++it) {
            const int c = tid + it * 256;
            const int r = c >> 2, q = c & 3;
            async_cp16(&xb[(size_t)(m0 + r) * I_ + k0 + q * 8], &Al[r * LDK + q * 8]);
        }
        // B tile: wih rows n0..n0+127 (already [n][k] layout)
        #pragma unroll
        for (int it = 0; it < 2; ++it) {
            const int c = tid + it * 256;
            const int r = c >> 2, q = c & 3;
            async_cp16(&wih[(size_t)(n0 + r) * I_ + k0 + q * 8], &Bl[r * LDK + q * 8]);
        }
        async_wait0();
        __syncthreads();

        v16bf a[4], b[2];
        #pragma unroll
        for (int mi = 0; mi < 4; ++mi)
            a[mi] = load_frag(&Al[(wm + mi * 16 + rsel) * LDK], ksel);
        #pragma unroll
        for (int ni = 0; ni < 2; ++ni)
            b[ni] = load_frag(&Bl[(wn + ni * 16 + rsel) * LDK], ksel);
        #pragma unroll
        for (int mi = 0; mi < 4; ++mi)
            #pragma unroll
            for (int ni = 0; ni < 2; ++ni)
                acc[mi][ni] = wmma_bf16(a[mi], b[ni], acc[mi][ni]);
    }

    const int mofs = (lane >> 4) * 8;
    #pragma unroll
    for (int mi = 0; mi < 4; ++mi) {
        #pragma unroll
        for (int ni = 0; ni < 2; ++ni) {
            const int gm = m0 + wm + mi * 16;
            const int gn = n0 + wn + ni * 16 + rsel;
            const float bv = bias[gn];
            #pragma unroll
            for (int r = 0; r < 8; ++r)
                Z[(size_t)(gm + r + mofs) * H_ + gn] = acc[mi][ni][r] + bv;
        }
    }
}

// ---------------------------------------------------------------------------
// Persistent recurrence kernel. Grid (16 n-tiles of 64, 2 directions).
// W_hh slice (64x1024 bf16) and full h_prev (64x1024 bf16) LDS-resident
// (258 KB < 320 KB WGP LDS). Device-scope atomic barrier between steps.
// Waves: 4(m) x 2(n), each 16x32 C strip -> 2 accumulators.
// ---------------------------------------------------------------------------
__global__ void __launch_bounds__(256) rnn_persistent_kernel(
    const __bf16* __restrict__ whh, const float* __restrict__ Z,
    __bf16* __restrict__ hb, float* __restrict__ out,
    unsigned* __restrict__ bar)
{
    __shared__ alignas(16) __bf16 Wl[64 * WST];
    __shared__ alignas(16) __bf16 Hl[64 * WST];

    const int dir  = blockIdx.y;
    const int n0   = blockIdx.x * 64;
    const int tid  = threadIdx.x;
    const int lane = tid & 31, wave = tid >> 5;
    const int wm   = (wave >> 1) * 16;   // 0,16,32,48 (batch rows)
    const int wnl  = (wave & 1) * 32;    // 0,32 (local cols)
    const int ksel = (lane >> 4) * 8;
    const int rsel = lane & 15;
    const int mofs = (lane >> 4) * 8;

    // Stage this WG's W_hh slice once: rows n0..n0+63, 128 chunks/row
    #pragma unroll
    for (int it = 0; it < 32; ++it) {
        const int c = tid + it * 256;
        const int r = c >> 7, q = c & 127;
        async_cp16(&whh[(size_t)(n0 + r) * H_ + q * 8], &Wl[r * WST + q * 8]);
    }
    async_wait0();
    __syncthreads();

    unsigned target = 0;
    for (int step = 0; step < S_; ++step) {
        const int t = dir ? (S_ - 1 - step) : step;
        const __bf16* hprev = hb + (size_t)(dir * 2 + (step & 1)) * B_ * H_;
        __bf16*       hnext = hb + (size_t)(dir * 2 + ((step + 1) & 1)) * B_ * H_;
        const float*  zrow  = Z + (size_t)t * B_ * H_;

        // Stage h_prev (64 x 1024 bf16)
        #pragma unroll
        for (int it = 0; it < 32; ++it) {
            const int c = tid + it * 256;
            const int r = c >> 7, q = c & 127;
            async_cp16(&hprev[(size_t)r * H_ + q * 8], &Hl[r * WST + q * 8]);
        }
        async_wait0();
        __syncthreads();

        v8f acc[2] = {};
        #pragma unroll 4
        for (int k0 = 0; k0 < H_; k0 += 32) {
            const v16bf a  = load_frag(&Hl[(wm + rsel) * WST + k0], ksel);
            const v16bf b0 = load_frag(&Wl[(wnl + rsel) * WST + k0], ksel);
            const v16bf b1 = load_frag(&Wl[(wnl + 16 + rsel) * WST + k0], ksel);
            acc[0] = wmma_bf16(a, b0, acc[0]);
            acc[1] = wmma_bf16(a, b1, acc[1]);
        }

        #pragma unroll
        for (int ni = 0; ni < 2; ++ni) {
            const int gn = n0 + wnl + ni * 16 + rsel;
            #pragma unroll
            for (int r = 0; r < 8; ++r) {
                const int m = wm + r + mofs;   // batch index
                const float v = tanhf(acc[ni][r] + zrow[(size_t)m * H_ + gn]);
                hnext[(size_t)m * H_ + gn] = (__bf16)v;
                if (step == S_ - 1)
                    out[(size_t)m * (2 * H_) + (size_t)dir * H_ + gn] = v;
            }
        }

        // Device-scope barrier across the 16 WGs of this direction
        target += 16;
        __threadfence();
        __syncthreads();
        if (tid == 0) {
            __hip_atomic_fetch_add(&bar[dir], 1u, __ATOMIC_RELEASE,
                                   __HIP_MEMORY_SCOPE_AGENT);
            while (__hip_atomic_load(&bar[dir], __ATOMIC_ACQUIRE,
                                     __HIP_MEMORY_SCOPE_AGENT) < target)
                __builtin_amdgcn_s_sleep(1);
        }
        __syncthreads();
    }
}

// ---------------------------------------------------------------------------
extern "C" void kernel_launch(void* const* d_in, const int* in_sizes, int n_in,
                              void* d_out, int out_size, void* d_ws, size_t ws_size,
                              hipStream_t stream) {
    (void)in_sizes; (void)n_in; (void)out_size; (void)ws_size;
    const float* x   = (const float*)d_in[0];
    const float* Wih = (const float*)d_in[1];
    const float* Whh = (const float*)d_in[2];
    const float* bih = (const float*)d_in[3];
    const float* bhh = (const float*)d_in[4];

    char* ws = (char*)d_ws;
    __bf16*   xb   = (__bf16*)(ws + XB_OFF);
    __bf16*   wihb = (__bf16*)(ws + WIH_OFF);
    __bf16*   whhb = (__bf16*)(ws + WHH_OFF);
    float*    bias = (float*)(ws + BIAS_OFF);
    __bf16*   hb   = (__bf16*)(ws + HB_OFF);   // [dir(2)][pp(2)][B_*H_]
    unsigned* bar  = (unsigned*)(ws + BAR_OFF);
    float*    Z    = (float*)(ws + Z_OFF);

    prep_kernel<<<2048, 256, 0, stream>>>(x, Wih, Whh, bih, bhh,
                                          xb, wihb, whhb, bias, hb, bar);

    zgemm_kernel<<<dim3((S_ * B_) / 128, H_ / 128), 256, 0, stream>>>(
        xb, wihb, bias, Z);

    rnn_persistent_kernel<<<dim3(H_ / 64, 2), 256, 0, stream>>>(
        whhb, Z, hb, (float*)d_out, bar);
}
